// HybridModel_81664508166391
// MI455X (gfx1250) — matrix operand order repaired and checked by reading, not compiled, not run on hardware
//
#include <hip/hip_runtime.h>
#include <hip/hip_bf16.h>

typedef __attribute__((ext_vector_type(16))) __bf16 v16bf;
typedef __attribute__((ext_vector_type(8)))  __bf16 v8bf;
typedef __attribute__((ext_vector_type(4)))  __bf16 v4bf;
typedef __attribute__((ext_vector_type(8)))  float  v8f;
typedef __attribute__((ext_vector_type(4)))  float  v4f;
typedef __attribute__((ext_vector_type(4)))  int    v4i;

#define FEAT   2048
#define NROWS  16384
#define NCLS   20
#define SMARG  0.31f

#define BM  128
#define BN  128
#define BK  32
#define LDT 56   // LDS row stride in bf16 elems (112B: 16B aligned, conflict-free)

// ---- CDNA5 async global->LDS path (guarded; sync fallback keeps compile safe) ----
#if defined(__has_builtin)
# if __has_builtin(__builtin_amdgcn_global_load_async_to_lds_b128)
#  define HAVE_ASYNC_LDS 1
# endif
# if __has_builtin(__builtin_amdgcn_s_wait_asynccnt)
#  define HAVE_WAIT_ASYNC 1
# endif
#endif

#define AS1 __attribute__((address_space(1)))
#define AS3 __attribute__((address_space(3)))

__device__ __forceinline__ AS3 v4i* lds_cast4(void* p) {
    // generic LDS address: low 32 bits are the LDS offset (ISA aperture rule)
    return (AS3 v4i*)(unsigned int)(unsigned long long)p;
}
__device__ __forceinline__ AS1 v4i* gbl_cast4(const void* p) {
    return (AS1 v4i*)(unsigned long long)p;
}
__device__ __forceinline__ void wait_async0() {
#if defined(HAVE_WAIT_ASYNC)
    __builtin_amdgcn_s_wait_asynccnt(0);
#else
    asm volatile("s_wait_asynccnt 0x0" ::: "memory");
#endif
}

// ---------------- fragment loaders ----------------
// A-matrix 16x32 bf16: lanes 0-15 hold K 0..15, lanes 16-31 K 16..31
__device__ __forceinline__ v16bf frag_a(const __bf16* p) {
    v8bf lo = *(const v8bf*)p;
    v8bf hi = *(const v8bf*)(p + 8);
    v16bf r;
#pragma unroll
    for (int i = 0; i < 8; ++i) { r[i] = lo[i]; r[i + 8] = hi[i]; }
    return r;
}
// B-matrix 32x16 bf16: lane = column N, 16 contiguous K (half-wave picks K 0-15 / 16-31)
__device__ __forceinline__ v16bf frag_b(const __bf16* p) {
    v8bf lo = *(const v8bf*)p;
    v8bf hi = *(const v8bf*)(p + 8);
    v16bf r;
#pragma unroll
    for (int i = 0; i < 8; ++i) { r[i] = lo[i]; r[i + 8] = hi[i]; }
    return r;
}

// ---------------- prep: W_enc^T -> bf16 [N][K] ----------------
__global__ void prep_wenc(const float* __restrict__ W, __bf16* __restrict__ wt) {
    unsigned i = blockIdx.x * blockDim.x + threadIdx.x;
    unsigned n = i >> 11, k = i & 2047;
    wt[i] = (__bf16)W[k * FEAT + n];
}

// ---------------- prep: [W_cls | pad | W_fl | pad]^T -> bf16 [64][K] ----------------
__global__ void prep_wcat(const float* __restrict__ Wc, const float* __restrict__ Wf,
                          __bf16* __restrict__ wcat) {
    unsigned i = blockIdx.x * blockDim.x + threadIdx.x;
    unsigned n = i >> 11, k = i & 2047;
    float v = 0.0f;
    if (n < NCLS)                      v = Wc[k * NCLS + n];
    else if (n >= 32 && n < 32 + NCLS) v = Wf[k * NCLS + (n - 32)];
    wcat[i] = (__bf16)v;
}

// ---------------- main GEMM: h = x @ W_enc + b_enc ----------------
struct SMem {
    union {
        struct { __bf16 a[2][BM * LDT]; __bf16 b[2][BN * LDT]; } ab;  // 56 KB
        __bf16 c[BM * BN];                                            // 32 KB
    };
};

__global__ __launch_bounds__(256, 1) void gemm_enc(
    const float* __restrict__ x, const __bf16* __restrict__ wt,
    const float* __restrict__ bias, __bf16* __restrict__ h) {
    __shared__ SMem sm;

    const int tid  = threadIdx.x;
    const int lane = tid & 31, wave = tid >> 5;
    const int wm = wave & 3, wn = wave >> 2;   // 4 waves along M, 2 along N
    const int lr = lane & 15, hi = lane >> 4;
    const int bm = blockIdx.y * BM, bn = blockIdx.x * BN;

    // per-thread staging coordinates
    const int arow = tid >> 3, akq = tid & 7;   // A: 32 rows/pass, 8 float4 per row
    const int brow = tid >> 2, bcq = tid & 3;   // B: 64 rows/pass, 4 chunks per row

    v8f acc[2][4];
#pragma unroll
    for (int i = 0; i < 2; ++i)
#pragma unroll
        for (int j = 0; j < 4; ++j)
#pragma unroll
            for (int e = 0; e < 8; ++e) acc[i][j][e] = 0.0f;

    // ---- staging helpers ----
    auto stageB = [&](int buf, int k0) {
#pragma unroll
        for (int it = 0; it < 2; ++it) {
            int row = brow + it * 64;
            const __bf16* src = wt + (size_t)(bn + row) * FEAT + k0 + bcq * 8;
            __bf16* dst = &sm.ab.b[buf][row * LDT + bcq * 8];
#if defined(HAVE_ASYNC_LDS)
            __builtin_amdgcn_global_load_async_to_lds_b128(
                gbl_cast4(src), lds_cast4(dst), 0, 0);
#else
            *(v8bf*)dst = *(const v8bf*)src;
#endif
        }
    };
    auto loadA = [&](int k0, v4f* areg) {
#pragma unroll
        for (int it = 0; it < 4; ++it) {
            int row = arow + it * 32;
            areg[it] = *(const v4f*)(x + (size_t)(bm + row) * FEAT + k0 + akq * 4);
        }
    };
    auto storeA = [&](int buf, const v4f* areg) {
#pragma unroll
        for (int it = 0; it < 4; ++it) {
            int row = arow + it * 32;
            v4bf b;
#pragma unroll
            for (int j = 0; j < 4; ++j) b[j] = (__bf16)areg[it][j];
            *(v4bf*)(&sm.ab.a[buf][row * LDT + akq * 4]) = b;
        }
    };

    // ---- prologue: fill buffer 0 ----
    {
        v4f areg[4];
        stageB(0, 0);
        loadA(0, areg);
        storeA(0, areg);
#if defined(HAVE_ASYNC_LDS)
        wait_async0();
#endif
    }
    __syncthreads();

    // ---- pipelined main loop: one barrier per k-step ----
    for (int k0 = 0; k0 < FEAT; k0 += BK) {
        const int cur  = (k0 >> 5) & 1;
        const int nxt  = cur ^ 1;
        const bool more = (k0 + BK) < FEAT;

        if (more) stageB(nxt, k0 + BK);          // async engine fills next B tile

        v16bf af[2], bfrag[4];
#pragma unroll
        for (int mt = 0; mt < 2; ++mt)
            af[mt] = frag_a(&sm.ab.a[cur][(wm * 32 + mt * 16 + lr) * LDT + hi * 16]);
#pragma unroll
        for (int nt = 0; nt < 4; ++nt)
            bfrag[nt] = frag_b(&sm.ab.b[cur][(wn * 64 + nt * 16 + lr) * LDT + hi * 16]);

        v4f areg[4];
        if (more) loadA(k0 + BK, areg);          // next A tile -> registers (LOADcnt)

#pragma unroll
        for (int mt = 0; mt < 2; ++mt)
#pragma unroll
            for (int nt = 0; nt < 4; ++nt)
                acc[mt][nt] = __builtin_amdgcn_wmma_f32_16x16x32_bf16(
                    false, af[mt], false, bfrag[nt], (short)0, acc[mt][nt],
                    false, false);

        if (more) {
            storeA(nxt, areg);                   // cvt + ds_store behind the WMMAs
#if defined(HAVE_ASYNC_LDS)
            wait_async0();
#endif
        }
        __syncthreads();
    }

    // ---- epilogue: +bias -> bf16 -> LDS stage -> coalesced b128 stores ----
#pragma unroll
    for (int mt = 0; mt < 2; ++mt)
#pragma unroll
        for (int nt = 0; nt < 4; ++nt) {
            int col  = wn * 64 + nt * 16 + lr;
            float bv = bias[bn + col];
            int row0 = wm * 32 + mt * 16 + hi * 8;
#pragma unroll
            for (int r = 0; r < 8; ++r)
                sm.c[(row0 + r) * BN + col] = (__bf16)(acc[mt][nt][r] + bv);
        }
    __syncthreads();
#pragma unroll
    for (int it = 0; it < 8; ++it) {
        int idx = it * 256 + tid;                // 2048 chunks of 8 bf16
        int row = idx >> 4, cq = idx & 15;
        v8bf v = *(const v8bf*)(&sm.c[row * BN + cq * 8]);
        *(v8bf*)(h + (size_t)(bm + row) * FEAT + bn + cq * 8) = v;
    }
}

// ---------------- head: logits = h @ [Wcls|Wfl], softmax / tau / mask ----------------
__global__ __launch_bounds__(128, 1) void head_kernel(
    const __bf16* __restrict__ h, const __bf16* __restrict__ wcat,
    const float* __restrict__ bc, const float* __restrict__ bf,
    float* __restrict__ out) {
    __shared__ float lg[4][16][64];
    const int lane = threadIdx.x & 31, wave = threadIdx.x >> 5;
    const int lr = lane & 15, hi = lane >> 4;
    const int row0 = blockIdx.x * 64 + wave * 16;

    v8f acc[4];
#pragma unroll
    for (int nt = 0; nt < 4; ++nt)
#pragma unroll
        for (int e = 0; e < 8; ++e) acc[nt][e] = 0.0f;

    for (int k0 = 0; k0 < FEAT; k0 += BK) {
        v16bf a = frag_a(h + (size_t)(row0 + lr) * FEAT + k0 + hi * 16);
#pragma unroll
        for (int nt = 0; nt < 4; ++nt) {
            v16bf b = frag_b(wcat + (size_t)(nt * 16 + lr) * FEAT + k0 + hi * 16);
            acc[nt] = __builtin_amdgcn_wmma_f32_16x16x32_bf16(
                false, a, false, b, (short)0, acc[nt], false, false);
        }
    }
#pragma unroll
    for (int nt = 0; nt < 4; ++nt)
#pragma unroll
        for (int r = 0; r < 8; ++r)
            lg[wave][hi * 8 + r][nt * 16 + lr] = acc[nt][r];
    __syncthreads();

    if (lane < 16) {
        const float* L = lg[wave][lane];
        float c[NCLS], f[NCLS];
        float m1 = -1e30f, m2 = -1e30f, fmin = 1e30f;
#pragma unroll
        for (int j = 0; j < NCLS; ++j) { c[j] = L[j] + bc[j]; m1 = fmaxf(m1, c[j]); }
#pragma unroll
        for (int j = 0; j < NCLS; ++j) {
            f[j] = L[32 + j] + bf[j];
            m2 = fmaxf(m2, f[j]); fmin = fminf(fmin, f[j]);
        }
        float s1 = 0.0f, s2 = 0.0f;
#pragma unroll
        for (int j = 0; j < NCLS; ++j) { c[j] = __expf(c[j] - m1); s1 += c[j]; }
#pragma unroll
        for (int j = 0; j < NCLS; ++j) s2 += __expf(f[j] - m2);
        float tau  = __expf(fmin - m2) / s2;   // min of flow softmax
        float pred = 1.0f / s1;                // max of cls softmax
        float scale = (pred >= tau + SMARG) ? (1.0f / s1) : 0.0f;
        int row = row0 + lane;
#pragma unroll
        for (int j = 0; j < NCLS; ++j)
            out[(size_t)row * NCLS + j] = c[j] * scale;
    }
}

// ---------------- launch ----------------
extern "C" void kernel_launch(void* const* d_in, const int* in_sizes, int n_in,
                              void* d_out, int out_size, void* d_ws, size_t ws_size,
                              hipStream_t stream) {
    const float* x     = (const float*)d_in[0];
    const float* W_enc = (const float*)d_in[1];
    const float* b_enc = (const float*)d_in[2];
    const float* W_cls = (const float*)d_in[3];
    const float* b_cls = (const float*)d_in[4];
    const float* W_fl  = (const float*)d_in[5];
    const float* b_fl  = (const float*)d_in[6];
    float* out = (float*)d_out;

    char* ws = (char*)d_ws;
    __bf16* wt   = (__bf16*)ws;                                        // 2048x2048 bf16 (8 MB)
    __bf16* wcat = (__bf16*)(ws + (size_t)FEAT * FEAT * 2);            // 64x2048 bf16
    __bf16* hbuf = (__bf16*)(ws + (size_t)FEAT * FEAT * 2
                                + (size_t)64 * FEAT * 2);              // 16384x2048 bf16 (64 MB)

    prep_wenc<<<dim3(FEAT * FEAT / 256), dim3(256), 0, stream>>>(W_enc, wt);
    prep_wcat<<<dim3(64 * FEAT / 256), dim3(256), 0, stream>>>(W_cls, W_fl, wcat);
    gemm_enc<<<dim3(FEAT / BN, NROWS / BM), dim3(256), 0, stream>>>(x, wt, b_enc, hbuf);
    head_kernel<<<dim3(NROWS / 64), dim3(128), 0, stream>>>(hbuf, wcat, b_cls, b_fl, out);
}
